// MultiScaleAttentionGraph_40114994545136
// MI455X (gfx1250) — matrix operand order repaired and checked by reading, not compiled, not run on hardware
//
#include <hip/hip_runtime.h>
#include <math.h>

// ---------------- types for WMMA (CDNA5 / gfx1250, wave32) ----------------
typedef __attribute__((ext_vector_type(16))) __bf16 v16bf;
typedef __attribute__((ext_vector_type(8)))  __bf16 v8bf;
typedef __attribute__((ext_vector_type(8)))  float  v8f;

#define SN0 100000
#define SN1 25000
#define SN2 6250
#define FDIM 128
#define INV_SCALE 0.17677669529663687f   // 1/sqrt(128/4)

// float -> bf16, round to nearest even
__device__ __forceinline__ __bf16 f2bf(float f) {
  unsigned u = __float_as_uint(f);
  unsigned r = (u + 0x7fffu + ((u >> 16) & 1u)) >> 16;
  unsigned short s = (unsigned short)r;
  return __builtin_bit_cast(__bf16, s);
}

// order-preserving float<->uint for atomicMax-based segment max
__device__ __forceinline__ unsigned fenc(float f) {
  unsigned u = __float_as_uint(f);
  return (u & 0x80000000u) ? ~u : (u | 0x80000000u);
}
__device__ __forceinline__ float fdec(unsigned k) {
  unsigned u = (k & 0x80000000u) ? (k & 0x7fffffffu) : ~k;
  return __uint_as_float(u);
}

// ---------------- fills ----------------
__global__ void fill_f32(float* p, float v, size_t n) {
  size_t i = (size_t)blockIdx.x * blockDim.x + threadIdx.x;
  size_t st = (size_t)gridDim.x * blockDim.x;
  for (; i < n; i += st) p[i] = v;
}
__global__ void fill_u32(unsigned* p, unsigned v, size_t n) {
  size_t i = (size_t)blockIdx.x * blockDim.x + threadIdx.x;
  size_t st = (size_t)gridDim.x * blockDim.x;
  for (; i < n; i += st) p[i] = v;
}

// ---------------- WMMA GEMM: C[n x 128] = (rowscale .* A)[n x 128] @ W[128 x 128] + bias --------
// block = 256 threads (8 wave32). Block computes a 128-row x 128-col stripe.
// Each wave owns one 16-row M-tile and sweeps all 8 N-tiles, reusing its A fragment
// across 8 WMMAs per K-step (32 WMMAs per wave). W is staged transposed in LDS as
// bf16 (sW[col][k]) so every B fragment is two contiguous ds_read_b128 per lane.
__global__ void __launch_bounds__(256)
gemm128_bias(const float* __restrict__ A, const float* __restrict__ W,
             const float* __restrict__ bias, const float* __restrict__ rowscale,
             float* __restrict__ C, int n) {
  __shared__ __bf16 sW[FDIM * FDIM];   // 32 KB, transposed: sW[col*128 + k]
  __shared__ __bf16 sA[FDIM * FDIM];   // 32 KB: sA[m*128 + k]
  const int tid = threadIdx.x;
  const int row0 = blockIdx.x * 128;

  // stage W transposed (b128 global reads; 4096 float4 / 256 threads = 16 iters)
  #pragma unroll 4
  for (int it = 0; it < 16; ++it) {
    int idx4 = it * 256 + tid;          // float4 index into W
    int k = idx4 >> 5;                  // 32 float4 per row of W
    int nb = (idx4 & 31) * 4;           // starting output-column
    float4 w = reinterpret_cast<const float4*>(W)[idx4];
    sW[(size_t)(nb + 0) * FDIM + k] = f2bf(w.x);
    sW[(size_t)(nb + 1) * FDIM + k] = f2bf(w.y);
    sW[(size_t)(nb + 2) * FDIM + k] = f2bf(w.z);
    sW[(size_t)(nb + 3) * FDIM + k] = f2bf(w.w);
  }
  // stage A tile (128 rows), optional row scaling, zero-pad tail rows
  #pragma unroll 4
  for (int it = 0; it < 16; ++it) {
    int idx4 = it * 256 + tid;
    int r  = idx4 >> 5;
    int kb = (idx4 & 31) * 4;
    int row = row0 + r;
    float4 v = make_float4(0.f, 0.f, 0.f, 0.f);
    float rs = 1.f;
    if (row < n) {
      v = reinterpret_cast<const float4*>(A + (size_t)row * FDIM)[idx4 & 31];
      if (rowscale) rs = rowscale[row];
    }
    __bf16* p = &sA[(size_t)r * FDIM + kb];
    p[0] = f2bf(v.x * rs); p[1] = f2bf(v.y * rs);
    p[2] = f2bf(v.z * rs); p[3] = f2bf(v.w * rs);
  }
  __syncthreads();

  const int wave = tid >> 5;      // M-tile index (0..7)
  const int lane = tid & 31;
  const int ln   = lane & 15;     // A: row M within tile, B: col N within tile
  const int half = lane >> 4;     // lane-half selects K sub-ranges
  const int klo  = half * 16;
  const int am   = wave * 16 + ln;

  const v8f vzero = {0.f,0.f,0.f,0.f,0.f,0.f,0.f,0.f};
  v8f acc[8];
  #pragma unroll
  for (int t = 0; t < 8; ++t) acc[t] = vzero;

  #pragma unroll
  for (int kk = 0; kk < FDIM; kk += 32) {
    // A fragment: lane<16 holds K kk..kk+15; lane>=16 holds kk+16..kk+31
    union { v16bf v; v8bf h[2]; } a;
    a.h[0] = *reinterpret_cast<const v8bf*>(&sA[(size_t)am * FDIM + kk + klo]);
    a.h[1] = *reinterpret_cast<const v8bf*>(&sA[(size_t)am * FDIM + kk + klo + 8]);
    const int kb = kk + half * 16;
    #pragma unroll
    for (int t = 0; t < 8; ++t) {
      // B fragment: lane<16 holds K kk..kk+15, lane>=16 holds kk+16..kk+31 (col = t*16+ln)
      v16bf b = *reinterpret_cast<const v16bf*>(&sW[(size_t)(t * 16 + ln) * FDIM + kb]);
      acc[t] = __builtin_amdgcn_wmma_f32_16x16x32_bf16(false, a.v, false, b,
                                                       (short)0, acc[t], false, false);
    }
  }

  #pragma unroll
  for (int t = 0; t < 8; ++t) {
    const int col = t * 16 + ln;
    const float bb = bias[col];
    #pragma unroll
    for (int r = 0; r < 8; ++r) {
      int grow = row0 + wave * 16 + r + half * 8;  // C layout: VGPR r -> M=r (+8 upper lanes)
      if (grow < n) C[(size_t)grow * FDIM + col] = acc[t][r] + bb;
    }
  }
}

// ---------------- score[i] = dot(K[i,:], q) * scale  (one wave per row) ----------------
__global__ void rowdot_kernel(const float* __restrict__ K, const float* __restrict__ q,
                              float* __restrict__ out, int n, float scale) {
  int wid = (int)(((size_t)blockIdx.x * blockDim.x + threadIdx.x) >> 5);
  int lane = threadIdx.x & 31;
  if (wid >= n) return;
  const float* row = K + (size_t)wid * FDIM;
  float s = 0.f;
  #pragma unroll
  for (int j = 0; j < 4; ++j) s += row[lane + 32 * j] * q[lane + 32 * j];
  #pragma unroll
  for (int off = 16; off; off >>= 1) s += __shfl_xor(s, off, 32);
  if (lane == 0) out[wid] = s * scale;
}

// ---------------- attention edge kernels ----------------
__global__ void edge_max_kernel(const int* __restrict__ src, const int* __restrict__ dst,
                                const float* __restrict__ score, unsigned* __restrict__ mkey, int E) {
  int e = (int)((size_t)blockIdx.x * blockDim.x + threadIdx.x);
  if (e >= E) return;
  atomicMax(&mkey[dst[e]], fenc(score[src[e]]));
}
__global__ void edge_exp_kernel(const int* __restrict__ src, const int* __restrict__ dst,
                                const float* __restrict__ score, const unsigned* __restrict__ mkey,
                                float* __restrict__ ew, float* __restrict__ denom, int E) {
  int e = (int)((size_t)blockIdx.x * blockDim.x + threadIdx.x);
  if (e >= E) return;
  int d = dst[e];
  float w = __expf(score[src[e]] - fdec(mkey[d]));
  ew[e] = w;
  atomicAdd(&denom[d], w);
}
// one wave per edge: out[dst] += (ew/denom[dst]) * V[src]
__global__ void attn_scatter_kernel(const int* __restrict__ src, const int* __restrict__ dst,
                                    const float* __restrict__ V, const float* __restrict__ ew,
                                    const float* __restrict__ denom, float* __restrict__ out, int E) {
  int e = (int)(((size_t)blockIdx.x * blockDim.x + threadIdx.x) >> 5);
  if (e >= E) return;
  int lane = threadIdx.x & 31;
  int s = src[e], d = dst[e];
  float a = ew[e] / denom[d];
  float4 v = reinterpret_cast<const float4*>(V + (size_t)s * FDIM)[lane];
  float* o = out + (size_t)d * FDIM + lane * 4;
  atomicAdd(o + 0, a * v.x); atomicAdd(o + 1, a * v.y);
  atomicAdd(o + 2, a * v.z); atomicAdd(o + 3, a * v.w);
}

// ---------------- GCN edge kernels ----------------
__global__ void edge_deg_kernel(const int* __restrict__ src, const int* __restrict__ dst,
                                float* __restrict__ dego, float* __restrict__ degi, int E) {
  int e = (int)((size_t)blockIdx.x * blockDim.x + threadIdx.x);
  if (e >= E) return;
  atomicAdd(&dego[src[e]], 1.f);
  atomicAdd(&degi[dst[e]], 1.f);
}
// one wave per edge: agg[dst] += rsqrt(max(dego[src],1)) * X[src]
__global__ void gcn_scatter_kernel(const int* __restrict__ src, const int* __restrict__ dst,
                                   const float* __restrict__ X, const float* __restrict__ dego,
                                   float* __restrict__ agg, int E) {
  int e = (int)(((size_t)blockIdx.x * blockDim.x + threadIdx.x) >> 5);
  if (e >= E) return;
  int lane = threadIdx.x & 31;
  int s = src[e], d = dst[e];
  float c = rsqrtf(fmaxf(dego[s], 1.f));
  float4 v = reinterpret_cast<const float4*>(X + (size_t)s * FDIM)[lane];
  float* o = agg + (size_t)d * FDIM + lane * 4;
  atomicAdd(o + 0, c * v.x); atomicAdd(o + 1, c * v.y);
  atomicAdd(o + 2, c * v.z); atomicAdd(o + 3, c * v.w);
}
__global__ void rsqrt_clip_kernel(float* p, int n) {
  int i = (int)((size_t)blockIdx.x * blockDim.x + threadIdx.x);
  if (i < n) p[i] = rsqrtf(fmaxf(p[i], 1.f));
}

// ---------------- head: softmax(concat(a,b) @ Wh + bh), 256 -> 5 ----------------
__global__ void head_kernel(const float* __restrict__ a, const float* __restrict__ b,
                            const float* __restrict__ Wh, const float* __restrict__ bh,
                            float* __restrict__ out, int n) {
  __shared__ float sW[256 * 5];
  __shared__ float sB[5];
  for (int i = threadIdx.x; i < 256 * 5; i += blockDim.x) sW[i] = Wh[i];
  if (threadIdx.x < 5) sB[threadIdx.x] = bh[threadIdx.x];
  __syncthreads();
  int i = (int)((size_t)blockIdx.x * blockDim.x + threadIdx.x);
  if (i >= n) return;
  float acc[5];
  #pragma unroll
  for (int o = 0; o < 5; ++o) acc[o] = sB[o];
  const float* ra = a + (size_t)i * FDIM;
  const float* rb = b + (size_t)i * FDIM;
  for (int k = 0; k < FDIM; ++k) {
    float xa = ra[k], xb = rb[k];
    #pragma unroll
    for (int o = 0; o < 5; ++o)
      acc[o] += xa * sW[k * 5 + o] + xb * sW[(128 + k) * 5 + o];
  }
  float m = acc[0];
  #pragma unroll
  for (int o = 1; o < 5; ++o) m = fmaxf(m, acc[o]);
  float s = 0.f;
  #pragma unroll
  for (int o = 0; o < 5; ++o) { acc[o] = __expf(acc[o] - m); s += acc[o]; }
  float inv = 1.f / s;
  #pragma unroll
  for (int o = 0; o < 5; ++o) out[(size_t)i * 5 + o] = acc[o] * inv;
}

// ---------------- orchestration ----------------
extern "C" void kernel_launch(void* const* d_in, const int* in_sizes, int n_in,
                              void* d_out, int out_size, void* d_ws, size_t ws_size,
                              hipStream_t stream) {
  (void)n_in; (void)out_size; (void)ws_size;
  const float* X      = (const float*)d_in[0];
  const float* emb_W  = (const float*)d_in[1];
  const float* emb_b  = (const float*)d_in[2];
  const float* gcn_W  = (const float*)d_in[3];
  const float* gcn_b  = (const float*)d_in[4];
  const float* enc_qb = (const float*)d_in[6];
  const float* enc_kW = (const float*)d_in[7];
  const float* enc_kb = (const float*)d_in[8];
  const float* enc_vW = (const float*)d_in[9];
  const float* enc_vb = (const float*)d_in[10];
  const float* dec_qb = (const float*)d_in[12];
  const float* dec_kW = (const float*)d_in[13];
  const float* dec_kb = (const float*)d_in[14];
  const float* dec_vW = (const float*)d_in[15];
  const float* dec_vb = (const float*)d_in[16];
  const float* head_W = (const float*)d_in[17];
  const float* head_b = (const float*)d_in[18];
  const int* g0s = (const int*)d_in[19]; const int* g0d = (const int*)d_in[20];
  const int* g1s = (const int*)d_in[21]; const int* g1d = (const int*)d_in[22];
  const int* g2s = (const int*)d_in[23]; const int* g2d = (const int*)d_in[24];
  const int* i0s = (const int*)d_in[25]; const int* i0d = (const int*)d_in[26];
  const int* i1s = (const int*)d_in[27]; const int* i1d = (const int*)d_in[28];
  const int* x0s = (const int*)d_in[29]; const int* x0d = (const int*)d_in[30];
  const int* x1s = (const int*)d_in[31]; const int* x1d = (const int*)d_in[32];
  const int E0 = in_sizes[19], E1 = in_sizes[21], E2 = in_sizes[23];
  const int Ei0 = in_sizes[25], Ei1 = in_sizes[27];
  const int Ex0 = in_sizes[29], Ex1 = in_sizes[31];
  float* outp = (float*)d_out;

  // workspace arena
  float* ws = (float*)d_ws;
  size_t off = 0;
  auto alloc = [&](size_t ne) { float* p = ws + off; off += ne; return p; };
  float* h0   = alloc((size_t)SN0 * FDIM);
  float* e1f  = alloc((size_t)SN1 * FDIM);
  float* e2f  = alloc((size_t)SN2 * FDIM);
  float* g0o  = alloc((size_t)SN0 * FDIM);
  float* g1o  = alloc((size_t)SN1 * FDIM);
  float* g2o  = alloc((size_t)SN2 * FDIM);
  float* d0b  = alloc((size_t)SN0 * FDIM);
  float* d1b  = alloc((size_t)SN1 * FDIM);
  float* Kb   = alloc((size_t)SN0 * FDIM);
  float* Vb   = alloc((size_t)SN0 * FDIM);
  float* agg  = alloc((size_t)SN0 * FDIM);
  float* score   = alloc(SN0);
  unsigned* mkey = (unsigned*)alloc(SN0);
  float* denom   = alloc(SN0);
  float* ew      = alloc(1000000);
  float* dego    = alloc(SN0);
  float* degi    = alloc(SN0);

  auto zf = [&](float* p, size_t n) {
    unsigned blks = (unsigned)((n + 255) / 256); if (blks > 4096u) blks = 4096u;
    fill_f32<<<blks, 256, 0, stream>>>(p, 0.f, n);
  };
  auto gemm = [&](const float* A, const float* W, const float* b, const float* rs, float* C, int n) {
    gemm128_bias<<<dim3((n + 127) / 128), dim3(256), 0, stream>>>(A, W, b, rs, C, n);
  };
  auto xattn = [&](const float* feat, int n_src, const int* es, const int* ed, int E, int n_dst,
                   const float* kW, const float* kb, const float* vW, const float* vb,
                   const float* qb, float* outbuf) {
    gemm(feat, kW, kb, nullptr, Kb, n_src);
    gemm(feat, vW, vb, nullptr, Vb, n_src);
    rowdot_kernel<<<dim3((n_src + 7) / 8), dim3(256), 0, stream>>>(Kb, qb, score, n_src, INV_SCALE);
    { unsigned blks = (unsigned)((n_dst + 255) / 256); if (blks > 4096u) blks = 4096u;
      fill_u32<<<blks, 256, 0, stream>>>(mkey, 0x007FFFFFu, (size_t)n_dst); } // enc(-inf)
    zf(denom, n_dst);
    zf(outbuf, (size_t)n_dst * FDIM);
    edge_max_kernel<<<dim3((E + 255) / 256), dim3(256), 0, stream>>>(es, ed, score, mkey, E);
    edge_exp_kernel<<<dim3((E + 255) / 256), dim3(256), 0, stream>>>(es, ed, score, mkey, ew, denom, E);
    attn_scatter_kernel<<<dim3((E + 7) / 8), dim3(256), 0, stream>>>(es, ed, Vb, ew, denom, outbuf, E);
  };
  auto gcn = [&](const float* feat, const int* es, const int* ed, int E, int n,
                 const float* W, const float* b, float* outbuf) {
    zf(dego, n); zf(degi, n); zf(agg, (size_t)n * FDIM);
    edge_deg_kernel<<<dim3((E + 255) / 256), dim3(256), 0, stream>>>(es, ed, dego, degi, E);
    gcn_scatter_kernel<<<dim3((E + 7) / 8), dim3(256), 0, stream>>>(es, ed, feat, dego, agg, E);
    rsqrt_clip_kernel<<<dim3((n + 255) / 256), dim3(256), 0, stream>>>(degi, n);
    gemm(agg, W, b, degi, outbuf, n);   // row-scale by D_in^-1/2 folded into GEMM
  };

  // 1) embedding
  gemm(X, emb_W, emb_b, nullptr, h0, SN0);
  // 2) encoder 0->1 and 1->2 (Q collapses to qb: score depends only on src)
  xattn(h0, SN0, i0s, i0d, Ei0, SN1,
        enc_kW, enc_kb, enc_vW, enc_vb, enc_qb, e1f);
  xattn(e1f, SN1, i1s, i1d, Ei1, SN2,
        enc_kW + 16384, enc_kb + 128, enc_vW + 16384, enc_vb + 128, enc_qb + 128, e2f);
  // 3) per-scale GCN
  gcn(h0,  g0s, g0d, E0, SN0, gcn_W,           gcn_b,       g0o);
  gcn(e1f, g1s, g1d, E1, SN1, gcn_W + 16384,   gcn_b + 128, g1o);
  gcn(e2f, g2s, g2d, E2, SN2, gcn_W + 32768,   gcn_b + 256, g2o);
  // 4) decoder 2->1 and 1->0
  xattn(g2o, SN2, x0s, x0d, Ex0, SN1,
        dec_kW, dec_kb, dec_vW, dec_vb, dec_qb, d1b);
  xattn(d1b, SN1, x1s, x1d, Ex1, SN0,
        dec_kW + 16384, dec_kb + 128, dec_vW + 16384, dec_vb + 128, dec_qb + 128, d0b);
  // 5) heads (decoded = [d0, d1, gcn_out2])
  head_kernel<<<dim3((SN0 + 255) / 256), dim3(256), 0, stream>>>(
      g0o, d0b, head_W,        head_b,      outp,                            SN0);
  head_kernel<<<dim3((SN1 + 255) / 256), dim3(256), 0, stream>>>(
      g1o, d1b, head_W + 1280, head_b + 5,  outp + (size_t)SN0 * 5,          SN1);
  head_kernel<<<dim3((SN2 + 255) / 256), dim3(256), 0, stream>>>(
      g2o, g2o, head_W + 2560, head_b + 10, outp + (size_t)(SN0 + SN1) * 5,  SN2);
}
